// LinearAttentionBlock_16793322127872
// MI455X (gfx1250) — matrix operand order repaired and checked by reading, not compiled, not run on hardware
//
#include <hip/hip_runtime.h>
#include <hip/hip_bf16.h>
#include <math.h>

// ---------------------------------------------------------------------------
// Problem constants (from reference): B=2, S=4096, D=1024, H=16, DH=64, FF=4096
// ---------------------------------------------------------------------------
#define BB 2
#define SS 4096
#define DD 1024
#define HH 16
#define DHH 64
#define FFD 4096
#define ROWS (BB * SS)          // 8192 token rows
#define NCHUNK 64               // S / 64 chunks per (b,h)
#define CHUNK 64

#define AS1 __attribute__((address_space(1)))
#define AS3 __attribute__((address_space(3)))

#if defined(__has_builtin)
#if __has_builtin(__builtin_amdgcn_global_load_async_to_lds_b128) && \
    __has_builtin(__builtin_amdgcn_s_wait_asynccnt)
#define HAVE_ASYNC_LDS 1
#endif
#endif

// GCC-vector int4: matches the builtin's V4i parameter type exactly.
typedef int v4i_vs __attribute__((__vector_size__(16)));

#ifdef HAVE_ASYNC_LDS
#define GASYNC_B128(gsrc, ldst)                                         \
  __builtin_amdgcn_global_load_async_to_lds_b128(                       \
      (AS1 v4i_vs*)(gsrc), (AS3 v4i_vs*)(ldst), 0, 0)
#endif

// ---------------------------------------------------------------------------
// bf16 helpers / WMMA types
// ---------------------------------------------------------------------------
typedef __bf16 bf16_t;
typedef bf16_t v16bf __attribute__((ext_vector_type(16)));
typedef float  v8f   __attribute__((ext_vector_type(8)));

struct alignas(16) U4 { unsigned int w[4]; };
union Frag { v16bf v; U4 u[2]; };

__device__ __forceinline__ unsigned short f2bf(float f) {
  union { float f; unsigned int u; } c; c.f = f;
  unsigned int u = c.u;
  unsigned int r = u + 0x7fffu + ((u >> 16) & 1u);   // RNE
  return (unsigned short)(r >> 16);
}

// ---------------------------------------------------------------------------
// Weight prep: wq' = wq @ blockdiag(proj), stored transposed (N x K) in bf16.
// ---------------------------------------------------------------------------
__global__ __launch_bounds__(256) void fold_proj_kernel(
    const float* __restrict__ w, const float* __restrict__ proj,
    unsigned short* __restrict__ outT) {
  int gid = blockIdx.x * 256 + threadIdx.x;       // 1024*1024 threads
  int n = gid >> 10, kd = gid & 1023;             // n = h*64 + j
  int h = n >> 6, j = n & 63;
  const float* wrow = w + (size_t)kd * DD + (h << 6);
  float s = 0.f;
#pragma unroll
  for (int d = 0; d < 64; ++d) s += wrow[d] * proj[d * 64 + j];
  outT[(size_t)n * DD + kd] = f2bf(s);
}

// Cast + transpose fp32 (K x N) -> bf16 (N x K)
__global__ __launch_bounds__(256) void cast_transpose_kernel(
    const float* __restrict__ w, unsigned short* __restrict__ outT,
    int K, int N) {
  int gid = blockIdx.x * 256 + threadIdx.x;       // K*N threads
  int n = gid / K, k = gid % K;
  outT[(size_t)n * K + k] = f2bf(w[(size_t)k * N + n]);
}

// ---------------------------------------------------------------------------
// LayerNorm fp32 row -> bf16 row
// ---------------------------------------------------------------------------
__global__ __launch_bounds__(256) void ln_to_bf16_kernel(
    const float* __restrict__ x, const float* __restrict__ g,
    const float* __restrict__ be, unsigned short* __restrict__ out) {
  __shared__ float r1[256], r2[256];
  int row = blockIdx.x, tid = threadIdx.x;
  const float* xr = x + (size_t)row * DD;
  float s = 0.f, s2 = 0.f;
#pragma unroll
  for (int i = 0; i < 4; ++i) {
    float v = xr[tid + i * 256];
    s += v; s2 += v * v;
  }
  r1[tid] = s; r2[tid] = s2;
  __syncthreads();
  for (int st = 128; st > 0; st >>= 1) {
    if (tid < st) { r1[tid] += r1[tid + st]; r2[tid] += r2[tid + st]; }
    __syncthreads();
  }
  float mean = r1[0] * (1.f / DD);
  float var  = r2[0] * (1.f / DD) - mean * mean;
  float rs   = rsqrtf(var + 1e-5f);
#pragma unroll
  for (int i = 0; i < 4; ++i) {
    int idx = tid + i * 256;
    out[(size_t)row * DD + idx] = f2bf((xr[idx] - mean) * rs * g[idx] + be[idx]);
  }
}

// ---------------------------------------------------------------------------
// bf16 WMMA GEMM: C(MxN, fp32 accum) = A(MxK bf16 row-major) * Bt(NxK bf16)ᵀ
// Block = 128 threads (4 waves), block tile 128x64; software-pipelined k-loop
// with register double buffering (8 WMMAs per 32-K step, 12 16B loads).
//
// Fragment layouts per gfx1250 ISA:
//   A 16x32 bf16: lane L holds row M=L&15; two contiguous 16B runs at
//                 K = k0+hi*8 and K = k0+16+hi*8        (hi = lane>=16)
//   B 32x16 bf16: lane L holds col N=L&15; two contiguous 16B runs at
//                 K = k0+hi*8 and K = k0+16+hi*8
// ---------------------------------------------------------------------------
#define GEMM_MODE_NONE      0   // outF = acc
#define GEMM_MODE_ELU1      1   // outF = elu(acc)+1
#define GEMM_MODE_BIAS_GELU 2   // outB = bf16(gelu_exact(acc + bias[n]))
#define GEMM_MODE_RES       3   // outF = acc + (bias?bias[n]:0) + res[m*N+n]

__global__ __launch_bounds__(128) void gemm_bf16_wmma_kernel(
    const unsigned short* __restrict__ A, const unsigned short* __restrict__ Bt,
    float* __restrict__ outF, unsigned short* __restrict__ outB,
    const float* __restrict__ bias, const float* __restrict__ res,
    int M, int N, int K, int mode) {
  const int lane = threadIdx.x & 31;
  const int wave = threadIdx.x >> 5;
  const int hi   = (lane >> 4) & 1;
  const int l15  = lane & 15;
  const int colBase = blockIdx.x * 64;

  v8f acc[2][4];
#pragma unroll
  for (int m = 0; m < 2; ++m)
#pragma unroll
    for (int t = 0; t < 4; ++t)
#pragma unroll
      for (int r = 0; r < 8; ++r) acc[m][t][r] = 0.f;

  const int rowA0 = blockIdx.y * 128 + wave * 32 + l15;
  const U4* arow0 = (const U4*)(A + (size_t)rowA0 * K);
  const U4* arow1 = (const U4*)(A + (size_t)(rowA0 + 16) * K);
  const U4* bro[4];
#pragma unroll
  for (int t = 0; t < 4; ++t)
    bro[t] = (const U4*)(Bt + (size_t)(colBase + t * 16 + l15) * K);

  auto loadFrags = [&](Frag* fa, Frag* fb, int k0) {
    const int kq = k0 >> 3;                    // uint4 index (8 bf16 / uint4)
    fa[0].u[0] = arow0[kq + hi];
    fa[0].u[1] = arow0[kq + 2 + hi];
    fa[1].u[0] = arow1[kq + hi];
    fa[1].u[1] = arow1[kq + 2 + hi];
#pragma unroll
    for (int t = 0; t < 4; ++t) {
      fb[t].u[0] = bro[t][kq + hi];
      fb[t].u[1] = bro[t][kq + 2 + hi];
    }
  };

  Frag fa0[2], fb0[4], fa1[2], fb1[4];
  loadFrags(fa0, fb0, 0);

  for (int k0 = 0; k0 < K; k0 += 64) {         // K is a multiple of 64
    loadFrags(fa1, fb1, k0 + 32);
#pragma unroll
    for (int m = 0; m < 2; ++m)
#pragma unroll
      for (int t = 0; t < 4; ++t)
        acc[m][t] = __builtin_amdgcn_wmma_f32_16x16x32_bf16(
            false, fa0[m].v, false, fb0[t].v, (short)0, acc[m][t], false, false);
    if (k0 + 64 < K) loadFrags(fa0, fb0, k0 + 64);
#pragma unroll
    for (int m = 0; m < 2; ++m)
#pragma unroll
      for (int t = 0; t < 4; ++t)
        acc[m][t] = __builtin_amdgcn_wmma_f32_16x16x32_bf16(
            false, fa1[m].v, false, fb1[t].v, (short)0, acc[m][t], false, false);
  }

  // Epilogue. C/D layout: lane L, vgpr r -> M = r + (L>=16 ? 8 : 0), N = L&15
#pragma unroll
  for (int m = 0; m < 2; ++m) {
    const int rbase = blockIdx.y * 128 + wave * 32 + m * 16 + hi * 8;
#pragma unroll
    for (int t = 0; t < 4; ++t) {
      const int col = colBase + t * 16 + l15;
#pragma unroll
      for (int r = 0; r < 8; ++r) {
        const int row = rbase + r;
        const size_t idx = (size_t)row * N + col;
        float a = acc[m][t][r];
        if (mode == GEMM_MODE_NONE) {
          outF[idx] = a;
        } else if (mode == GEMM_MODE_ELU1) {
          outF[idx] = (a > 0.f) ? (a + 1.f) : __expf(a);
        } else if (mode == GEMM_MODE_BIAS_GELU) {
          float tv = a + bias[col];
          float gl = 0.5f * tv * (1.f + erff(tv * 0.70710678118654752f));
          outB[idx] = f2bf(gl);
        } else {  // GEMM_MODE_RES
          float bz = bias ? bias[col] : 0.f;
          outF[idx] = a + bz + res[idx];
        }
      }
    }
  }
}

// ---------------------------------------------------------------------------
// Chunked linear attention (layout: ((b*S+s)*H + h)*DH + d).
// Pass 1: per-(b,h,chunk) KVc = k^T v, ksum = colsum(k)
// Scan  : exclusive prefix over chunk axis
// Pass 2: out_t = (q_t@KVprev + sum_{j<=t}(q_t.k_j) v_j) / (den + 1e-6)
// LDS tiles staged via gfx1250 async global->LDS copies when available.
// ---------------------------------------------------------------------------
__global__ __launch_bounds__(64) void attn_pass1_kernel(
    const float* __restrict__ k, const float* __restrict__ v,
    float* __restrict__ KVc, float* __restrict__ ksc) {
  __shared__ float kS[64][64];
  __shared__ float vS[64][64];
  const int blk = blockIdx.x;           // bh*64 + c
  const int bh = blk >> 6, c = blk & 63;
  const int b = bh >> 4, h = bh & 15;
  const int tid = threadIdx.x;
  const size_t base = (((size_t)b * SS + (size_t)c * 64) * HH + h) * DHH;
#ifdef HAVE_ASYNC_LDS
  for (int idx = tid * 4; idx < 4096; idx += 64 * 4) {
    int j = idx >> 6, d = idx & 63;
    size_t g = base + (size_t)j * (HH * DHH) + d;
    GASYNC_B128(k + g, &kS[j][d]);
    GASYNC_B128(v + g, &vS[j][d]);
  }
  __builtin_amdgcn_s_wait_asynccnt(0);
#else
  for (int idx = tid; idx < 4096; idx += 64) {
    int j = idx >> 6, d = idx & 63;
    size_t g = base + (size_t)j * (HH * DHH) + d;
    kS[j][d] = k[g];
    vS[j][d] = v[g];
  }
#endif
  __syncthreads();
  float kv[64];
#pragma unroll
  for (int e = 0; e < 64; ++e) kv[e] = 0.f;
  float ks = 0.f;
  for (int j = 0; j < 64; ++j) {
    float kd = kS[j][tid];
    ks += kd;
#pragma unroll
    for (int e = 0; e < 64; ++e) kv[e] += kd * vS[j][e];
  }
  size_t o = ((size_t)blk * 64 + tid) * 64;
#pragma unroll
  for (int e = 0; e < 64; ++e) KVc[o + e] = kv[e];
  ksc[(size_t)blk * 64 + tid] = ks;
}

__global__ __launch_bounds__(256) void attn_scan_kernel(
    float* __restrict__ KVc, float* __restrict__ ksc) {
  const int bh = blockIdx.x;            // 32 blocks
  const int tid = threadIdx.x;
  for (int comp = tid; comp < 4096; comp += 256) {
    float run = 0.f;
    for (int c = 0; c < NCHUNK; ++c) {
      size_t a = ((size_t)(bh * NCHUNK + c)) * 4096 + comp;
      float t = KVc[a];
      KVc[a] = run;                     // exclusive prefix
      run += t;
    }
  }
  if (tid < 64) {
    float run = 0.f;
    for (int c = 0; c < NCHUNK; ++c) {
      size_t a = (size_t)(bh * NCHUNK + c) * 64 + tid;
      float t = ksc[a];
      ksc[a] = run;
      run += t;
    }
  }
}

__global__ __launch_bounds__(64) void attn_pass2_kernel(
    const float* __restrict__ q, const float* __restrict__ k,
    const float* __restrict__ v, const float* __restrict__ KVc,
    const float* __restrict__ ksc, unsigned short* __restrict__ attnb) {
  __shared__ float kS[64][64];
  __shared__ float vS[64][64];
  __shared__ float kvS[64][64];
  __shared__ float ksumS[64];
  const int blk = blockIdx.x;
  const int bh = blk >> 6, c = blk & 63;
  const int b = bh >> 4, h = bh & 15;
  const int tid = threadIdx.x;
  const size_t base = (((size_t)b * SS + (size_t)c * 64) * HH + h) * DHH;
#ifdef HAVE_ASYNC_LDS
  for (int idx = tid * 4; idx < 4096; idx += 64 * 4) {
    int j = idx >> 6, d = idx & 63;
    size_t g = base + (size_t)j * (HH * DHH) + d;
    GASYNC_B128(k + g, &kS[j][d]);
    GASYNC_B128(v + g, &vS[j][d]);
    GASYNC_B128(KVc + (size_t)blk * 4096 + idx, &kvS[j][d]);
  }
  if (tid < 16)
    GASYNC_B128(ksc + (size_t)blk * 64 + tid * 4, &ksumS[tid * 4]);
  __builtin_amdgcn_s_wait_asynccnt(0);
#else
  for (int idx = tid; idx < 4096; idx += 64) {
    int j = idx >> 6, d = idx & 63;
    size_t g = base + (size_t)j * (HH * DHH) + d;
    kS[j][d] = k[g];
    vS[j][d] = v[g];
    kvS[j][d] = KVc[(size_t)blk * 4096 + idx];
  }
  if (tid < 64) ksumS[tid] = ksc[(size_t)blk * 64 + tid];
#endif
  __syncthreads();

  float qr[64];
  const size_t qbase = base + (size_t)tid * (HH * DHH);
#pragma unroll
  for (int d = 0; d < 64; ++d) qr[d] = q[qbase + d];

  float num[64];
#pragma unroll
  for (int e = 0; e < 64; ++e) num[e] = 0.f;
  float den = 1e-6f;

  // inter-chunk: q_t @ KV_prev, q_t . ksum_prev
#pragma unroll 4
  for (int d = 0; d < 64; ++d) {
    float qd = qr[d];
    den += qd * ksumS[d];
#pragma unroll
    for (int e = 0; e < 64; ++e) num[e] += qd * kvS[d][e];
  }
  // intra-chunk causal part (includes diagonal)
  for (int j = 0; j <= tid; ++j) {
    float s = 0.f;
#pragma unroll
    for (int d = 0; d < 64; ++d) s += qr[d] * kS[j][d];
    den += s;
#pragma unroll
    for (int e = 0; e < 64; ++e) num[e] += s * vS[j][e];
  }
  float inv = 1.f / den;
  const size_t ob = ((size_t)b * SS + (size_t)c * 64 + tid) * DD + (size_t)h * DHH;
#pragma unroll
  for (int e = 0; e < 64; ++e) attnb[ob + e] = f2bf(num[e] * inv);
}

// ---------------------------------------------------------------------------
// Host-side orchestration
// ---------------------------------------------------------------------------
extern "C" void kernel_launch(void* const* d_in, const int* in_sizes, int n_in,
                              void* d_out, int out_size, void* d_ws, size_t ws_size,
                              hipStream_t stream) {
  const float* x     = (const float*)d_in[0];
  const float* wq    = (const float*)d_in[1];
  const float* wk    = (const float*)d_in[2];
  const float* wv    = (const float*)d_in[3];
  const float* wo    = (const float*)d_in[4];
  const float* proj  = (const float*)d_in[5];
  const float* w1    = (const float*)d_in[6];
  const float* b1    = (const float*)d_in[7];
  const float* w2    = (const float*)d_in[8];
  const float* b2    = (const float*)d_in[9];
  const float* g1    = (const float*)d_in[10];
  const float* beta1 = (const float*)d_in[11];
  const float* g2    = (const float*)d_in[12];
  const float* beta2 = (const float*)d_in[13];
  float* out = (float*)d_out;

  // Workspace carve-up (256B aligned)
  char* ws = (char*)d_ws;
  size_t off = 0;
  auto carve = [&](size_t bytes) -> char* {
    char* p = ws + off;
    off += (bytes + 255) & ~(size_t)255;
    return p;
  };
  unsigned short* h_bf  = (unsigned short*)carve((size_t)ROWS * DD * 2);
  unsigned short* wqfT  = (unsigned short*)carve((size_t)DD * DD * 2);
  unsigned short* wkfT  = (unsigned short*)carve((size_t)DD * DD * 2);
  unsigned short* wvT   = (unsigned short*)carve((size_t)DD * DD * 2);
  unsigned short* woT   = (unsigned short*)carve((size_t)DD * DD * 2);
  unsigned short* w1T   = (unsigned short*)carve((size_t)DD * FFD * 2);
  unsigned short* w2T   = (unsigned short*)carve((size_t)FFD * DD * 2);
  float* qf   = (float*)carve((size_t)ROWS * DD * 4);
  float* kf   = (float*)carve((size_t)ROWS * DD * 4);
  float* vf   = (float*)carve((size_t)ROWS * DD * 4);
  float* KVc  = (float*)carve((size_t)BB * HH * NCHUNK * DHH * DHH * 4);
  float* ksc  = (float*)carve((size_t)BB * HH * NCHUNK * DHH * 4);
  unsigned short* attnb = (unsigned short*)carve((size_t)ROWS * DD * 2);
  float* x2   = (float*)carve((size_t)ROWS * DD * 4);
  unsigned short* h2_bf = (unsigned short*)carve((size_t)ROWS * DD * 2);
  unsigned short* midb  = (unsigned short*)carve((size_t)ROWS * FFD * 2);

  // 1) Weight prep (fold proj into wq/wk; transpose+cast the rest)
  fold_proj_kernel<<<(DD * DD) / 256, 256, 0, stream>>>(wq, proj, wqfT);
  fold_proj_kernel<<<(DD * DD) / 256, 256, 0, stream>>>(wk, proj, wkfT);
  cast_transpose_kernel<<<(DD * DD) / 256, 256, 0, stream>>>(wv, wvT, DD, DD);
  cast_transpose_kernel<<<(DD * DD) / 256, 256, 0, stream>>>(wo, woT, DD, DD);
  cast_transpose_kernel<<<(DD * FFD) / 256, 256, 0, stream>>>(w1, w1T, DD, FFD);
  cast_transpose_kernel<<<(FFD * DD) / 256, 256, 0, stream>>>(w2, w2T, FFD, DD);

  // 2) LN1 -> bf16
  ln_to_bf16_kernel<<<ROWS, 256, 0, stream>>>(x, g1, beta1, h_bf);

  // 3) Projections (WMMA): q = elu1(h@wq'), k = elu1(h@wk'), v = h@wv
  dim3 gemmB(128);
  dim3 gD(DD / 64, ROWS / 128);
  gemm_bf16_wmma_kernel<<<gD, gemmB, 0, stream>>>(h_bf, wqfT, qf, nullptr,
      nullptr, nullptr, ROWS, DD, DD, GEMM_MODE_ELU1);
  gemm_bf16_wmma_kernel<<<gD, gemmB, 0, stream>>>(h_bf, wkfT, kf, nullptr,
      nullptr, nullptr, ROWS, DD, DD, GEMM_MODE_ELU1);
  gemm_bf16_wmma_kernel<<<gD, gemmB, 0, stream>>>(h_bf, wvT, vf, nullptr,
      nullptr, nullptr, ROWS, DD, DD, GEMM_MODE_NONE);

  // 4) Chunked linear attention
  attn_pass1_kernel<<<BB * HH * NCHUNK, 64, 0, stream>>>(kf, vf, KVc, ksc);
  attn_scan_kernel<<<BB * HH, 256, 0, stream>>>(KVc, ksc);
  attn_pass2_kernel<<<BB * HH * NCHUNK, 64, 0, stream>>>(qf, kf, vf, KVc, ksc, attnb);

  // 5) Output projection + residual: x2 = x + attn @ wo
  gemm_bf16_wmma_kernel<<<gD, gemmB, 0, stream>>>(attnb, woT, x2, nullptr,
      nullptr, x, ROWS, DD, DD, GEMM_MODE_RES);

  // 6) LN2 -> bf16
  ln_to_bf16_kernel<<<ROWS, 256, 0, stream>>>(x2, g2, beta2, h2_bf);

  // 7) FFN: mid = gelu(h2@w1 + b1) (bf16), out = x2 + mid@w2 + b2
  dim3 gF(FFD / 64, ROWS / 128);
  gemm_bf16_wmma_kernel<<<gF, gemmB, 0, stream>>>(h2_bf, w1T, nullptr, midb,
      b1, nullptr, ROWS, FFD, DD, GEMM_MODE_BIAS_GELU);
  gemm_bf16_wmma_kernel<<<gD, gemmB, 0, stream>>>(midb, w2T, out, nullptr,
      b2, x2, ROWS, DD, FFD, GEMM_MODE_RES);
}
